// GAT_27659589386644
// MI455X (gfx1250) — compile-verified, matrix operand order checked by reading, not run on hardware
//
#include <hip/hip_runtime.h>

// ---------------- CDNA5 (gfx1250, wave32) WMMA types ----------------
typedef __attribute__((ext_vector_type(16))) __bf16 v16bf;
typedef __attribute__((ext_vector_type(8)))  __bf16 v8bf;
typedef __attribute__((ext_vector_type(8)))  float  v8f;

union BF16x16 { v16bf v; v8bf h[2]; };
union BF16x8  { v8bf  v; __bf16 e[8]; };

__device__ __forceinline__ float bf2f(__bf16 x) {
  unsigned short u = __builtin_bit_cast(unsigned short, x);
  unsigned int   w = ((unsigned int)u) << 16;
  return __builtin_bit_cast(float, w);
}
__device__ __forceinline__ __bf16 f2bf(float f) {
  unsigned int u = __builtin_bit_cast(unsigned int, f);
  u += 0x7FFFu + ((u >> 16) & 1u);  // round to nearest even
  unsigned short s = (unsigned short)(u >> 16);
  return __builtin_bit_cast(__bf16, s);
}

#define GB 16      // batch
#define GN 256     // nodes
#define GE 4       // edge types
#define GH 8       // heads
#define GD 128     // dim
#define GEH (GE*GH)     // 32
#define GBN (GB*GN)     // 4096
#define GDIN1 (GEH*GD)  // 4096

// ---------------- embedding gather -> bf16 state0 ----------------
__global__ __launch_bounds__(256) void gat_embed(const int* __restrict__ nf,
                                                 const float* __restrict__ emb,
                                                 __bf16* __restrict__ out) {
  int i = blockIdx.x * 256 + threadIdx.x;       // GBN*GD
  int node = i >> 7, d = i & 127;
  out[i] = f2bf(emb[nf[node] * GD + d]);
}

// ---------------- W0 (E1,H,D,D) -> bf16 [eh][o][i] ----------------
__global__ __launch_bounds__(256) void gat_cvt_w0(const float* __restrict__ W,
                                                  __bf16* __restrict__ Wt) {
  int idx = blockIdx.x * 256 + threadIdx.x;     // GEH*GD*GD
  int eh = idx >> 14;
  int o  = (idx >> 7) & 127;
  int i  = idx & 127;
  Wt[idx] = f2bf(W[((size_t)eh * GD + i) * GD + o]);
}

// ---------------- W1 (E1,H,4096,D) -> bf16 [eh][o][i] ----------------
__global__ __launch_bounds__(256) void gat_cvt_w1(const float* __restrict__ W,
                                                  __bf16* __restrict__ Wt) {
  size_t idx = (size_t)blockIdx.x * 256 + threadIdx.x;  // GEH*GDIN1*GD
  int eh = (int)(idx >> 19);
  int i  = (int)(idx & 4095);
  // dst layout: idx = eh<<19 | o<<12 | i
  Wt[idx] = f2bf(W[((size_t)eh * GDIN1 + i) * GD + ((int)((idx >> 12) & 127))]);
}

// ------------- LDS-tiled WMMA GEMM: Out^T[eh][d][m] = X[m][k] * Wt[eh][d][k] -------------
// Block: 256 threads / 8 waves -> 256(M) x 128(N) per eh.  Double-buffered LDS staging:
// As[2][256][40], Bs[2][128][40] bf16 (pad 40 kills ds_load_b128 bank conflicts).
#define TLD 40
__global__ __launch_bounds__(256) void gat_gemm(const __bf16* __restrict__ X,
                                                const __bf16* __restrict__ Wt,
                                                __bf16* __restrict__ OutT,
                                                int K) {
  const int eh   = blockIdx.z;
  const int tid  = threadIdx.x;
  const int wave = tid >> 5, lane = tid & 31, l15 = lane & 15, hi = lane >> 4;
  const int mrow0 = blockIdx.x * 256;

  __shared__ __align__(16) __bf16 As[2][256 * TLD];
  __shared__ __align__(16) __bf16 Bs[2][128 * TLD];

  const __bf16* Xb = X + (size_t)mrow0 * K;
  const __bf16* Wb = Wt + (size_t)eh * GD * K;

  // A: 256 rows x 32 k = 1024 16B-chunks; thread does 4. B: 128 x 32 = 512 chunks; thread does 2.
  v8bf areg[4], breg[2];

  auto fetch = [&](int k0) {
#pragma unroll
    for (int c = 0; c < 4; ++c) {
      int chunk = c * 256 + tid;
      int row = chunk >> 2, sub = chunk & 3;
      areg[c] = *(const v8bf*)(Xb + (size_t)row * K + k0 + sub * 8);
    }
#pragma unroll
    for (int c = 0; c < 2; ++c) {
      int chunk = c * 256 + tid;
      int col = chunk >> 2, sub = chunk & 3;
      breg[c] = *(const v8bf*)(Wb + (size_t)col * K + k0 + sub * 8);
    }
  };
  auto stash = [&](int buf) {
#pragma unroll
    for (int c = 0; c < 4; ++c) {
      int chunk = c * 256 + tid;
      int row = chunk >> 2, sub = chunk & 3;
      *(v8bf*)&As[buf][row * TLD + sub * 8] = areg[c];
    }
#pragma unroll
    for (int c = 0; c < 2; ++c) {
      int chunk = c * 256 + tid;
      int col = chunk >> 2, sub = chunk & 3;
      *(v8bf*)&Bs[buf][col * TLD + sub * 8] = breg[c];
    }
  };

  v8f acc[2][8];
#pragma unroll
  for (int mt = 0; mt < 2; ++mt)
#pragma unroll
    for (int nt = 0; nt < 8; ++nt) acc[mt][nt] = (v8f){};

  const int nk = K >> 5;
  fetch(0);
  stash(0);

  for (int ks = 0; ks < nk; ++ks) {
    const int cur = ks & 1;
    if (ks + 1 < nk) fetch((ks + 1) << 5);  // prefetch next tile into regs
    __syncthreads();                        // cur buffer fully written

    BF16x16 A[2];
#pragma unroll
    for (int mt = 0; mt < 2; ++mt) {
      const int row = (wave * 2 + mt) * 16 + l15;
      const __bf16* ap = &As[cur][row * TLD + hi * 8];
      A[mt].h[0] = *(const v8bf*)ap;        // ds_load_b128
      A[mt].h[1] = *(const v8bf*)(ap + 16);
    }
#pragma unroll
    for (int nt = 0; nt < 8; ++nt) {
      const __bf16* bp = &Bs[cur][(nt * 16 + l15) * TLD + hi * 8];
      BF16x16 Bv;
      Bv.h[0] = *(const v8bf*)bp;
      Bv.h[1] = *(const v8bf*)(bp + 16);
#pragma unroll
      for (int mt = 0; mt < 2; ++mt)
        acc[mt][nt] = __builtin_amdgcn_wmma_f32_16x16x32_bf16(
            false, A[mt].v, false, Bv.v, (short)0, acc[mt][nt], false, false);
    }
    __syncthreads();                        // everyone done reading before refill
    if (ks + 1 < nk) stash(1 - cur);
  }

  // transposed store: OutT[eh][d][m]; lane holds column d, 8 consecutive rows
#pragma unroll
  for (int mt = 0; mt < 2; ++mt) {
    const int mbase = mrow0 + (wave * 2 + mt) * 16 + hi * 8;
#pragma unroll
    for (int nt = 0; nt < 8; ++nt) {
      const int d = nt * 16 + l15;
      BF16x8 ov;
#pragma unroll
      for (int r = 0; r < 8; ++r) ov.e[r] = f2bf(acc[mt][nt][r]);
      *(v8bf*)(OutT + ((size_t)eh * GD + d) * GBN + mbase) = ov.v;
    }
  }
}

// ------------- s1/s2: s[eh][m] = sum_d WhT[eh][d][m]*aw[eh][d] + ab[eh] -------------
__global__ __launch_bounds__(256) void gat_s1s2(const __bf16* __restrict__ WhT,
                                                const float* __restrict__ a1w,
                                                const float* __restrict__ a1b,
                                                const float* __restrict__ a2w,
                                                const float* __restrict__ a2b,
                                                float* __restrict__ s1,
                                                float* __restrict__ s2) {
  int i  = blockIdx.x * 256 + threadIdx.x;   // GEH*GBN, threads cover consecutive m
  int eh = i >> 12, m = i & 4095;
  const __bf16* p = WhT + (size_t)eh * GD * GBN + m;
  float r1 = 0.f, r2 = 0.f;
  for (int d = 0; d < GD; ++d) {
    float v = bf2f(p[(size_t)d * GBN]);
    r1 += v * a1w[eh * GD + d];
    r2 += v * a2w[eh * GD + d];
  }
  s1[i] = r1 + a1b[eh];
  s2[i] = r2 + a2b[eh];
}

__global__ __launch_bounds__(256) void gat_zero(float* __restrict__ p) {
  p[blockIdx.x * 256 + threadIdx.x] = 0.f;
}

// ------------- fused attention: col-softmax(logits) @ Wh, per (e,h,b) block -------------
// logits[n][m] = lrelu(s1[n]+s2[m],0.2) + L[b][n][m][e]; softmax over n (column-wise).
// LAYER 0: elu(h + sb) -> state1 bf16 [b*n][eh*128+d]
// LAYER 1: (h + sb)/32 atomically accumulated into hF [b*n][d]
#define ATT_LD 40
template <int LAYER>
__global__ __launch_bounds__(256) void gat_att(const float* __restrict__ s1g,
                                               const float* __restrict__ s2g,
                                               const float* __restrict__ Lmat,
                                               const __bf16* __restrict__ WhT,
                                               const float* __restrict__ sb,
                                               __bf16* __restrict__ state1,
                                               float* __restrict__ hF) {
  const int b = blockIdx.x, h = blockIdx.y, e = blockIdx.z;
  const int eh  = e * GH + h;
  const int tid = threadIdx.x;

  __shared__ float s1s[GN], s2s[GN], cmax[GN], csum[GN];
  __shared__ __align__(16) __bf16 att[GN * ATT_LD];

  s1s[tid] = s1g[((size_t)eh * GB + b) * GN + tid];
  s2s[tid] = s2g[((size_t)eh * GB + b) * GN + tid];
  __syncthreads();

  // streaming column softmax stats, column m = tid
  {
    const int m = tid;
    const float s2v = s2s[m];
    const float* Lp = Lmat + (((size_t)b * GN) * GN + m) * GE + e;  // n stride = GN*GE
    float mx = -3.0e38f, sm = 0.f;
    for (int n = 0; n < GN; ++n) {
      float v = s1s[n] + s2v;
      v = v > 0.f ? v : 0.2f * v;
      v += Lp[(size_t)n * (GN * GE)];
      float nm = fmaxf(mx, v);
      sm = sm * __expf(mx - nm) + __expf(v - nm);
      mx = nm;
    }
    cmax[m] = mx;
    csum[m] = sm;
  }
  __syncthreads();

  const int wave = tid >> 5, lane = tid & 31, l15 = lane & 15, hi = lane >> 4;
  v8f acc[2][8];
#pragma unroll
  for (int mt = 0; mt < 2; ++mt)
#pragma unroll
    for (int nt = 0; nt < 8; ++nt) acc[mt][nt] = (v8f){};

  const size_t whbase = (size_t)eh * GD * GBN + (size_t)b * GN;  // WhT[eh][d][b*GN + m]

  for (int kb = 0; kb < 8; ++kb) {          // 8 blocks of 32 source-columns m
    {  // build bf16 att tile [n][mm], n = tid
      const int n = tid;
      const float s1v = s1s[n];
      const float* Lq = Lmat + (((size_t)b * GN + n) * GN + kb * 32) * GE + e;
      for (int mm = 0; mm < 32; ++mm) {
        const int m = kb * 32 + mm;
        float v = s1v + s2s[m];
        v = v > 0.f ? v : 0.2f * v;
        v += Lq[mm * GE];
        att[n * ATT_LD + mm] = f2bf(__expf(v - cmax[m]) / csum[m]);
      }
    }
    __syncthreads();

    BF16x16 A[2];
#pragma unroll
    for (int mt = 0; mt < 2; ++mt) {
      const int row = (wave * 2 + mt) * 16 + l15;
      const __bf16* ap = &att[row * ATT_LD + hi * 8];
      A[mt].h[0] = *(const v8bf*)ap;          // ds_load_b128
      A[mt].h[1] = *(const v8bf*)(ap + 16);
    }
#pragma unroll
    for (int nt = 0; nt < 8; ++nt) {
      const int d = nt * 16 + l15;
      const __bf16* bp = WhT + whbase + (size_t)d * GBN + kb * 32 + hi * 8;
      BF16x16 Bv;
      Bv.h[0] = *(const v8bf*)bp;
      Bv.h[1] = *(const v8bf*)(bp + 16);
#pragma unroll
      for (int mt = 0; mt < 2; ++mt)
        acc[mt][nt] = __builtin_amdgcn_wmma_f32_16x16x32_bf16(
            false, A[mt].v, false, Bv.v, (short)0, acc[mt][nt], false, false);
    }
    __syncthreads();
  }

  // epilogue
#pragma unroll
  for (int mt = 0; mt < 2; ++mt) {
    const int mbase = (wave * 2 + mt) * 16 + hi * 8;
#pragma unroll
    for (int nt = 0; nt < 8; ++nt) {
      const int d = nt * 16 + l15;
      const float sbv = sb[eh * GD + d];
#pragma unroll
      for (int r = 0; r < 8; ++r) {
        const int n = mbase + r;
        float v = acc[mt][nt][r] + sbv;
        if (LAYER == 0) {
          v = v > 0.f ? v : __expf(v) - 1.f;  // elu
          state1[((size_t)b * GN + n) * GDIN1 + (size_t)eh * GD + d] = f2bf(v);
        } else {
          atomicAdd(&hF[((size_t)b * GN + n) * GD + d], v * (1.0f / GEH));
        }
      }
    }
  }
}

// ------------- readout: y = stateF@outW+outb; aw = sigmoid(stateF@attW+attb); mean_n -------------
__global__ __launch_bounds__(128) void gat_final(const float* __restrict__ hF,
                                                 const float* __restrict__ outW,
                                                 const float* __restrict__ outb,
                                                 const float* __restrict__ attW,
                                                 const float* __restrict__ attb,
                                                 float* __restrict__ out) {
  const int b = blockIdx.x, d = threadIdx.x;
  __shared__ float row[GD], red[GD];
  float acc = 0.f;
  for (int n = 0; n < GN; ++n) {
    row[d] = hF[((size_t)b * GN + n) * GD + d];
    __syncthreads();
    float y = outb[d];
    for (int i = 0; i < GD; ++i) y += row[i] * outW[i * GD + d];
    red[d] = row[d] * attW[d];
    __syncthreads();
    for (int s = 64; s > 0; s >>= 1) {
      if (d < s) red[d] += red[d + s];
      __syncthreads();
    }
    const float aw = 1.f / (1.f + __expf(-(red[0] + attb[0])));
    acc += aw * y;
    __syncthreads();
  }
  out[b * GD + d] = acc * (1.0f / GN);
}

// ---------------------------------------------------------------------------
extern "C" void kernel_launch(void* const* d_in, const int* in_sizes, int n_in,
                              void* d_out, int out_size, void* d_ws, size_t ws_size,
                              hipStream_t stream) {
  const int*   nf    = (const int*)d_in[0];
  const float* L     = (const float*)d_in[1];
  const float* emb   = (const float*)d_in[2];
  const float* W0    = (const float*)d_in[3];
  const float* W1    = (const float*)d_in[4];
  const float* a1w0  = (const float*)d_in[5];
  const float* a1b0  = (const float*)d_in[6];
  const float* a2w0  = (const float*)d_in[7];
  const float* a2b0  = (const float*)d_in[8];
  const float* a1w1  = (const float*)d_in[9];
  const float* a1b1  = (const float*)d_in[10];
  const float* a2w1  = (const float*)d_in[11];
  const float* a2b1  = (const float*)d_in[12];
  const float* sb0   = (const float*)d_in[13];
  const float* sb1   = (const float*)d_in[14];
  const float* outW  = (const float*)d_in[15];
  const float* outb  = (const float*)d_in[16];
  const float* attW  = (const float*)d_in[17];
  const float* attb  = (const float*)d_in[18];
  float* out = (float*)d_out;

  // workspace carve-up (256B aligned)
  size_t off = 0;
  auto carve = [&](size_t bytes) {
    void* p = (char*)d_ws + off;
    off += (bytes + 255) & ~(size_t)255;
    return p;
  };
  __bf16* state0  = (__bf16*)carve((size_t)GBN * GD * 2);            //   1 MB
  __bf16* W0t     = (__bf16*)carve((size_t)GEH * GD * GD * 2);       //   1 MB
  __bf16* W1t     = (__bf16*)carve((size_t)GEH * GD * GDIN1 * 2);    //  33.5 MB
  __bf16* WhT     = (__bf16*)carve((size_t)GEH * GD * GBN * 2);      //  33.5 MB
  __bf16* state1  = (__bf16*)carve((size_t)GBN * GDIN1 * 2);         //  33.5 MB
  float*  s1      = (float*)carve((size_t)GEH * GBN * 4);            //   0.5 MB
  float*  s2      = (float*)carve((size_t)GEH * GBN * 4);            //   0.5 MB
  float*  hF      = (float*)carve((size_t)GBN * GD * 4);             //   2 MB
  (void)ws_size;

  const dim3 blk256(256), blk128(128);

  // layer 0
  gat_embed<<<(GBN * GD) / 256, blk256, 0, stream>>>(nf, emb, state0);
  gat_cvt_w0<<<(GEH * GD * GD) / 256, blk256, 0, stream>>>(W0, W0t);
  gat_cvt_w1<<<(int)(((size_t)GEH * GD * GDIN1) / 256), blk256, 0, stream>>>(W1, W1t);
  gat_gemm<<<dim3(GBN / 256, 1, GEH), blk256, 0, stream>>>(state0, W0t, WhT, GD);
  gat_s1s2<<<(GEH * GBN) / 256, blk256, 0, stream>>>(WhT, a1w0, a1b0, a2w0, a2b0, s1, s2);
  gat_att<0><<<dim3(GB, GH, GE), blk256, 0, stream>>>(s1, s2, L, WhT, sb0, state1, nullptr);

  // layer 1 (dominant: 32 GEMMs of 4096x4096x128 via WMMA bf16)
  gat_gemm<<<dim3(GBN / 256, 1, GEH), blk256, 0, stream>>>(state1, W1t, WhT, GDIN1);
  gat_s1s2<<<(GEH * GBN) / 256, blk256, 0, stream>>>(WhT, a1w1, a1b1, a2w1, a2b1, s1, s2);
  gat_zero<<<(GBN * GD) / 256, blk256, 0, stream>>>(hF);
  gat_att<1><<<dim3(GB, GH, GE), blk256, 0, stream>>>(s1, s2, L, WhT, sb1, nullptr, hF);

  // readout
  gat_final<<<GB, blk128, 0, stream>>>(hF, outW, outb, attW, attb, out);
}
